// Attention_77506979824044
// MI455X (gfx1250) — compile-verified, matrix operand order checked by reading, not run on hardware
//
#include <hip/hip_runtime.h>
#include <hip/hip_bf16.h>

// B=8, T=2048, C=1024, H=64. fp32 in/out, bf16 WMMA internally (fp32 accum).
#define TDIM 2048
#define CDIM 1024
#define HDIM 64
#define NBATCH 8

typedef __attribute__((ext_vector_type(16))) __bf16 v16bf;
typedef __attribute__((ext_vector_type(8)))  float  v8f;

// CDNA5 16-bit fragment packing (wave32): lane = half*16 + idx.
// A-frag: row = idx, halfs 0..7 hold K = k0 + half*8 + j, halfs 8..15 hold K = k0+16+half*8+j.
// B-frag: column = idx, same K packing. Both load contiguous-K rows identically.
__device__ __forceinline__ v16bf frag_from_f32(const float* __restrict__ row, int k0, int half) {
    v16bf f;
    const float* p0 = row + k0 + half * 8;
    const float* p1 = row + k0 + 16 + half * 8;
#pragma unroll
    for (int j = 0; j < 8; ++j) { f[j] = (__bf16)p0[j]; f[j + 8] = (__bf16)p1[j]; }
    return f;
}

__device__ __forceinline__ v16bf frag_from_bf16(const __bf16* __restrict__ row, int k0, int half) {
    v16bf f;
    const __bf16* p0 = row + k0 + half * 8;        // 8 bf16 = 16B -> b128 load
    const __bf16* p1 = row + k0 + 16 + half * 8;
#pragma unroll
    for (int j = 0; j < 8; ++j) { f[j] = p0[j]; f[j + 8] = p1[j]; }
    return f;
}

__device__ __forceinline__ v8f wmma_bf16(v16bf a, v16bf b, v8f c) {
    return __builtin_amdgcn_wmma_f32_16x16x32_bf16(false, a, false, b, (short)0, c, false, false);
}

// ---------------- Kernel 0: one-time W fp32 -> bf16 conversion ----------------
// Weights are loop-invariant across all 1024 projection waves; convert once so
// the hot loop streams bf16 (half the bytes, no per-step v_cvt on weights).
__global__ __launch_bounds__(256) void cvt_w_kernel(
    const float* __restrict__ Wq, const float* __restrict__ Wk,
    const float* __restrict__ Wv, __bf16* __restrict__ Wb)
{
    const int n = HDIM * CDIM;                     // 65536 per matrix
    const int i = blockIdx.x * 256 + threadIdx.x;
    if (i < n) {
        Wb[i]         = (__bf16)Wq[i];
        Wb[i + n]     = (__bf16)Wk[i];
        Wb[i + 2 * n] = (__bf16)Wv[i];
    }
}

// ---------------- Kernel 1: fused QKV projection ----------------
// One wave per 16-row strip of x; reuses the A fragment across 12 output tiles
// (Q/K/V x 4 head-tiles). Stores Q (pre-scaled by C^-0.5), K row-major bf16,
// and V transposed ([64][T] bf16) for the P@V B-fragments later.
__global__ __launch_bounds__(128) void qkv_proj_kernel(
    const float* __restrict__ x, const __bf16* __restrict__ Wb,
    __bf16* __restrict__ Qb, __bf16* __restrict__ Kb, __bf16* __restrict__ Vt)
{
    const int lane = threadIdx.x & 31;
    const int idx  = lane & 15;
    const int half = lane >> 4;
    const int strip = blockIdx.x * 4 + (threadIdx.x >> 5);   // 0 .. B*T/16 - 1
    const size_t m0 = (size_t)strip * 16;
    const float* xrow = x + (m0 + idx) * CDIM;

    const __bf16* Wqb = Wb;
    const __bf16* Wkb = Wb + HDIM * CDIM;
    const __bf16* Wvb = Wb + 2 * HDIM * CDIM;

    v8f accQ[4] = {}, accK[4] = {}, accV[4] = {};

    for (int k = 0; k < CDIM; k += 32) {
        v16bf a = frag_from_f32(xrow, k, half);
#pragma unroll
        for (int nt = 0; nt < 4; ++nt) {
            const int wrow = (nt * 16 + idx) * CDIM;         // B column n = W row n
            accQ[nt] = wmma_bf16(a, frag_from_bf16(Wqb + wrow, k, half), accQ[nt]);
            accK[nt] = wmma_bf16(a, frag_from_bf16(Wkb + wrow, k, half), accK[nt]);
            accV[nt] = wmma_bf16(a, frag_from_bf16(Wvb + wrow, k, half), accV[nt]);
        }
    }

    const float qscale = 0.03125f;  // C^-0.5 = 1024^-0.5, folded into Q
#pragma unroll
    for (int nt = 0; nt < 4; ++nt) {
#pragma unroll
        for (int i = 0; i < 8; ++i) {
            const size_t m = m0 + i + 8 * half;       // C layout: row = i + 8*half
            const int    n = nt * 16 + idx;           //           col = idx
            Qb[m * HDIM + n] = (__bf16)(accQ[nt][i] * qscale);
            Kb[m * HDIM + n] = (__bf16)(accK[nt][i]);
            const size_t b = m >> 11;                 // m / 2048
            const size_t t = m & (TDIM - 1);
            Vt[(b * HDIM + n) * TDIM + t] = (__bf16)(accV[nt][i]);
        }
    }
}

// ---------------- Kernel 2: causal flash attention ----------------
// One wave per 16-query tile. Computes S^T = K_tile @ Q^T so that softmax
// reductions are in-lane (+ one shfl_xor(16)) and exp(S^T) registers are
// directly the A-fragment for the P@V WMMA — no LDS transpose.
__global__ __launch_bounds__(128) void flash_attn_kernel(
    const __bf16* __restrict__ Qb, const __bf16* __restrict__ Kb,
    const __bf16* __restrict__ Vt, float* __restrict__ out)
{
    const int lane = threadIdx.x & 31;
    const int idx  = lane & 15;
    const int half = lane >> 4;
    const int wid  = blockIdx.x * 4 + (threadIdx.x >> 5);    // 0 .. B*(T/16)-1
    const int b    = wid >> 7;                               // 128 q-tiles / batch
    const int q0   = (wid & 127) * 16;

    const __bf16* Qrow  = Qb + ((size_t)b * TDIM + q0) * HDIM;
    const __bf16* Kbase = Kb + (size_t)b * TDIM * HDIM;
    const __bf16* Vbase = Vt + (size_t)b * HDIM * TDIM;

    // Q^T B-fragments (column q = Q row q, contiguous along head): loop invariant.
    const v16bf qf0 = frag_from_bf16(Qrow + idx * HDIM, 0, half);
    const v16bf qf1 = frag_from_bf16(Qrow + idx * HDIM, 32, half);

    v8f accO[4] = {};
    float mrun = -1e30f, lrun = 0.f;
    const int qg = q0 + idx;                                 // this lane-column's query

    for (int c0 = 0; c0 < q0 + 16; c0 += 32) {               // 32-key chunks up to diagonal
        const __bf16* Krow0 = Kbase + (size_t)(c0 + idx) * HDIM;
        const __bf16* Krow1 = Kbase + (size_t)(c0 + 16 + idx) * HDIM;
        v8f s0 = {}, s1 = {};
        s0 = wmma_bf16(frag_from_bf16(Krow0, 0, half), qf0, s0);
        s0 = wmma_bf16(frag_from_bf16(Krow0, 32, half), qf1, s0);
        s1 = wmma_bf16(frag_from_bf16(Krow1, 0, half), qf0, s1);
        s1 = wmma_bf16(frag_from_bf16(Krow1, 32, half), qf1, s1);

        // S^T C layout: component i, lane(half,idx) = S[key = c0 + i + 8*half (+16)][query = q0+idx]
        float p0[8], p1[8];
        float mt = -1e30f;
#pragma unroll
        for (int i = 0; i < 8; ++i) {
            const int kg = c0 + 8 * half + i;
            const float a0 = (kg <= qg) ? s0[i] : -1e30f;          // causal mask
            const float a1 = (kg + 16 <= qg) ? s1[i] : -1e30f;
            p0[i] = a0; p1[i] = a1;
            mt = fmaxf(mt, fmaxf(a0, a1));
        }
        mt = fmaxf(mt, __shfl_xor(mt, 16, 32));                    // combine halves
        const float mnew  = fmaxf(mrun, mt);
        const float alpha = __expf(mrun - mnew);
        mrun = mnew;

        float rsum = 0.f;
        v16bf pf;   // A-fragment of P for P@V: halfs 0..7 = tile0, 8..15 = tile1 (in-lane!)
#pragma unroll
        for (int i = 0; i < 8; ++i) {
            const float e0 = __expf(p0[i] - mnew);
            const float e1 = __expf(p1[i] - mnew);
            rsum += e0 + e1;
            pf[i]     = (__bf16)e0;
            pf[i + 8] = (__bf16)e1;
        }
        rsum += __shfl_xor(rsum, 16, 32);
        lrun = lrun * alpha + rsum;

        // Rescale O: alpha lives per lane-column (q = idx); O needs it per component
        // (q = i + 8*half). Gather from lane 8*half + i.
        float al[8];
#pragma unroll
        for (int i = 0; i < 8; ++i) al[i] = __shfl(alpha, 8 * half + i, 32);
#pragma unroll
        for (int nt = 0; nt < 4; ++nt)
#pragma unroll
            for (int i = 0; i < 8; ++i) accO[nt][i] *= al[i];

        // O += P(16x32) @ V_chunk(32x16 per head-tile); B column h = Vt row h (contiguous keys).
#pragma unroll
        for (int nt = 0; nt < 4; ++nt) {
            v16bf vf = frag_from_bf16(Vbase + (size_t)(nt * 16 + idx) * TDIM, c0, half);
            accO[nt] = wmma_bf16(pf, vf, accO[nt]);
        }
    }

    float li[8];
#pragma unroll
    for (int i = 0; i < 8; ++i) li[i] = __shfl(lrun, 8 * half + i, 32);

    float* orow = out + ((size_t)b * TDIM + q0) * HDIM;
#pragma unroll
    for (int nt = 0; nt < 4; ++nt)
#pragma unroll
        for (int i = 0; i < 8; ++i)
            orow[(size_t)(i + 8 * half) * HDIM + nt * 16 + idx] = accO[nt][i] / li[i];
}

extern "C" void kernel_launch(void* const* d_in, const int* in_sizes, int n_in,
                              void* d_out, int out_size, void* d_ws, size_t ws_size,
                              hipStream_t stream) {
    const float* x  = (const float*)d_in[0];
    const float* Wq = (const float*)d_in[1];
    const float* Wk = (const float*)d_in[2];
    const float* Wv = (const float*)d_in[3];
    float* out = (float*)d_out;

    const size_t nQKV = (size_t)NBATCH * TDIM * HDIM;   // 1M elems each
    __bf16* wsQ  = (__bf16*)d_ws;
    __bf16* wsK  = wsQ + nQKV;
    __bf16* wsVt = wsK + nQKV;                          // 6 MB bf16 Q/K/V^T
    __bf16* wsW  = wsVt + nQKV;                         // + 384 KB bf16 weights

    // 64*1024 = 65536 elems per weight matrix; one thread each.
    cvt_w_kernel<<<256, 256, 0, stream>>>(Wq, Wk, Wv, wsW);
    // B*T/16 = 1024 strips / waves; 4 waves per block over 256 blocks (wider WGP spread).
    qkv_proj_kernel<<<256, 128, 0, stream>>>(x, wsW, wsQ, wsK, wsVt);
    // B*(T/16) = 1024 query tiles / waves; same spread.
    flash_attn_kernel<<<256, 128, 0, stream>>>(wsQ, wsK, wsVt, out);
}
